// Memorynet_81990925680885
// MI455X (gfx1250) — compile-verified, matrix operand order checked by reading, not run on hardware
//
#include <hip/hip_runtime.h>
#include <hip/hip_bf16.h>

// ---------------------------------------------------------------------------
// Memorynet forward for gfx1250 (MI455X).  All dense math runs through
// v_wmma_f32_16x16x32_bf16 (fp32 data cast to bf16, fp32 accumulate).
// B matrices (weights) are stored transposed (N x Kpad) so both A and B
// fragments are two contiguous 16-byte LDS reads per lane (ds_load_b128).
// k_gemm uses 32x32 per-wave tiles: 4 WMMA per 4 fragment loads per K-step.
// ---------------------------------------------------------------------------

typedef __bf16 bf16;
typedef bf16  v16bf __attribute__((ext_vector_type(16)));
typedef float v8f   __attribute__((ext_vector_type(8)));

#define DEV __device__ __forceinline__

static constexpr int BB = 4;        // batch
static constexpr int N0 = 8192;     // input points
static constexpr int KNB = 32;      // neighbors per ball query
static constexpr float FINF = 3.0e38f;

DEV unsigned short f2bf(float f) {                 // fp32 -> bf16 (RNE)
  unsigned u = __builtin_bit_cast(unsigned, f);
  unsigned r = 0x7FFFu + ((u >> 16) & 1u);
  return (unsigned short)((u + r) >> 16);
}

// Build a 16-element bf16 fragment from a K-contiguous 32-element tile row.
// Wave32 layout (ISA 7.12.2): lanes 0-15 take K{0..7,16..23},
// lanes 16-31 take K{8..15,24..31}.  Two uint4 (16B) LDS loads per lane.
DEV v16bf frag_tile(const unsigned short* tile, int idx0, int lane) {
  int half = lane >> 4;
  const unsigned short* base = tile + (idx0 + (lane & 15)) * 32;
  struct V { uint4 a, b; } v;
  v.a = *(const uint4*)(base + (half ? 8 : 0));
  v.b = *(const uint4*)(base + (half ? 24 : 16));
  return __builtin_bit_cast(v16bf, v);
}

// ---------------------------------------------------------------------------
// WMMA GEMM + BN + ReLU + max over groups of 32 rows (ball-query pooling).
// A: (groups*32, Kpad) bf16 row-major    Bt: (N, Kpad) bf16 (transposed W)
// out: (groups, N) fp32.  grid = (groups, N/128), block = 256
// 8 waves = 2 Mtiles x 4 Npairs; each wave does 2 wmma (16x32 of N) per step.
// ---------------------------------------------------------------------------
__global__ void __launch_bounds__(256)
k_gemm_pool(const unsigned short* __restrict__ A, const unsigned short* __restrict__ Bt,
            const float* __restrict__ g, const float* __restrict__ beta,
            float* __restrict__ out, int Kpad, int N) {
  __shared__ unsigned short As[32 * 32];    // 32 rows  x 32 k
  __shared__ unsigned short Bs[128 * 32];   // 128 cols x 32 k (K-contiguous)
  __shared__ float pool[256];

  int t = threadIdx.x, lane = t & 31, w = t >> 5;
  int mt = w >> 2, nt = w & 3;
  int group = blockIdx.x;
  int ncol0 = blockIdx.y * 128;
  const unsigned short* Arow = A + (long)group * 32 * Kpad;

  v8f c0, c1;
#pragma unroll
  for (int v = 0; v < 8; ++v) { c0[v] = 0.f; c1[v] = 0.f; }

  for (int k0 = 0; k0 < Kpad; k0 += 32) {
    {  // stage A tile 32x32: 1024 elems, one uint2 per thread
      int e = t * 4, r = e >> 5, cc = e & 31;
      *(uint2*)(&As[e]) = *(const uint2*)(Arow + (long)r * Kpad + k0 + cc);
    }
#pragma unroll
    for (int j = 0; j < 2; ++j) {  // stage Bt tile 128x32: two uint4 per thread
      int e = t * 8 + j * 2048, col = e >> 5, kk = e & 31;
      *(uint4*)(&Bs[e]) = *(const uint4*)(Bt + (long)(ncol0 + col) * Kpad + k0 + kk);
    }
    if (k0 + 32 < Kpad)
      __builtin_prefetch(Bt + (long)(ncol0 + lane * 4) * Kpad + k0 + 32, 0, 0);
    __syncthreads();
    v16bf a  = frag_tile(As, mt * 16, lane);
    v16bf b0 = frag_tile(Bs, nt * 32, lane);
    v16bf b1 = frag_tile(Bs, nt * 32 + 16, lane);
    c0 = __builtin_amdgcn_wmma_f32_16x16x32_bf16(false, a, false, b0, (short)0, c0, false, false);
    c1 = __builtin_amdgcn_wmma_f32_16x16x32_bf16(false, a, false, b1, (short)0, c1, false, false);
    __syncthreads();
  }

#pragma unroll
  for (int h = 0; h < 2; ++h) {
    v8f& c = h ? c1 : c0;
    int col = ncol0 + nt * 32 + h * 16 + (lane & 15);
    float gam = g[col], bet = beta[col];
    float m8 = -FINF;
#pragma unroll
    for (int v = 0; v < 8; ++v) m8 = fmaxf(m8, fmaxf(c[v] * gam + bet, 0.f)); // bn+relu, then max
    float m16 = fmaxf(m8, __shfl_xor(m8, 16, 32));
    if ((lane >> 4) == 0) pool[mt * 128 + nt * 32 + h * 16 + lane] = m16;
  }
  __syncthreads();
  if (t < 128) out[(long)group * N + ncol0 + t] = fmaxf(pool[t], pool[128 + t]);
}

// ---------------------------------------------------------------------------
// Generic WMMA GEMM: out = [relu]( A@B * g + beta [+ res] )
// A: (M, K) bf16 row-major, Bt: (N, K) bf16 (transposed weight), K%32==0,
// N%64==0, M%128==0.  grid = (M/128, N/64), block 256.
// 8 waves = 4 Mquads x 2 Npairs; each wave owns a 32x32 tile: 4 wmma from
// 2 A-fragments + 2 B-fragments per K step (max reuse per LDS byte).
// ---------------------------------------------------------------------------
__global__ void __launch_bounds__(256)
k_gemm(const unsigned short* __restrict__ A, const unsigned short* __restrict__ Bt,
       const float* __restrict__ g, const float* __restrict__ beta,
       const float* __restrict__ res, float* __restrict__ out,
       int Kdim, int N, int relu) {
  __shared__ unsigned short As[128 * 32];   // 8 KB
  __shared__ unsigned short Bs[64 * 32];    // 4 KB

  int t = threadIdx.x, lane = t & 31, w = t >> 5;
  int mt = w >> 1, nt = w & 1;
  long row0 = (long)blockIdx.x * 128;
  int ncol0 = blockIdx.y * 64;

  v8f c00, c01, c10, c11;
#pragma unroll
  for (int v = 0; v < 8; ++v) { c00[v] = 0.f; c01[v] = 0.f; c10[v] = 0.f; c11[v] = 0.f; }

  for (int k0 = 0; k0 < Kdim; k0 += 32) {
#pragma unroll
    for (int j = 0; j < 2; ++j) {  // A tile 128x32: 4096 elems, two uint4/thread
      int e = t * 8 + j * 2048, r = e >> 5, cc = e & 31;
      *(uint4*)(&As[e]) = *(const uint4*)(A + (row0 + r) * Kdim + k0 + cc);
    }
    {                              // Bt tile 64x32: 2048 elems, one uint4/thread
      int e = t * 8, col = e >> 5, kk = e & 31;
      *(uint4*)(&Bs[e]) = *(const uint4*)(Bt + (long)(ncol0 + col) * Kdim + k0 + kk);
    }
    if (k0 + 32 < Kdim)
      __builtin_prefetch(Bt + (long)(ncol0 + lane * 2) * Kdim + k0 + 32, 0, 0);
    __syncthreads();
    v16bf a0 = frag_tile(As, mt * 32, lane);
    v16bf a1 = frag_tile(As, mt * 32 + 16, lane);
    v16bf b0 = frag_tile(Bs, nt * 32, lane);
    v16bf b1 = frag_tile(Bs, nt * 32 + 16, lane);
    c00 = __builtin_amdgcn_wmma_f32_16x16x32_bf16(false, a0, false, b0, (short)0, c00, false, false);
    c01 = __builtin_amdgcn_wmma_f32_16x16x32_bf16(false, a0, false, b1, (short)0, c01, false, false);
    c10 = __builtin_amdgcn_wmma_f32_16x16x32_bf16(false, a1, false, b0, (short)0, c10, false, false);
    c11 = __builtin_amdgcn_wmma_f32_16x16x32_bf16(false, a1, false, b1, (short)0, c11, false, false);
    __syncthreads();
  }

  int rsub = 8 * (lane >> 4);
#pragma unroll
  for (int ms = 0; ms < 2; ++ms) {
#pragma unroll
    for (int h = 0; h < 2; ++h) {
      v8f& c = ms ? (h ? c11 : c10) : (h ? c01 : c00);
      int col = ncol0 + nt * 32 + h * 16 + (lane & 15);
      float gam = g ? g[col] : 1.f;
      float bet = beta ? beta[col] : 0.f;
#pragma unroll
      for (int v = 0; v < 8; ++v) {
        long row = row0 + mt * 32 + ms * 16 + rsub + v;
        float val = c[v] * gam + bet;
        if (res) val += res[row * N + col];
        if (relu) val = fmaxf(val, 0.f);
        out[row * N + col] = val;
      }
    }
  }
}

// ------------------------------ scalar kernels -----------------------------

__global__ void k_in_linear(const float* __restrict__ feat, const float* __restrict__ w,
                            const float* __restrict__ bias, float* __restrict__ x, long total) {
  long i = (long)blockIdx.x * blockDim.x + threadIdx.x;
  if (i >= total) return;
  long row = i >> 6; int c = i & 63;
  const float* f = feat + row * 4;
  float acc = bias[c];
#pragma unroll
  for (int k = 0; k < 4; ++k) acc += f[k] * w[k * 64 + c];
  x[i] = acc;
}

// farthest point sampling; one block per batch, dist cache in LDS.
__global__ void k_fps(const float* __restrict__ pts, int n, int m, int* __restrict__ idx) {
  extern __shared__ float dis[];
  __shared__ float rv[256];
  __shared__ int ri[256];
  __shared__ int s_last;
  const float* base = pts + (long)blockIdx.x * n * 3;
  int t = threadIdx.x;
  for (int i = t; i < n; i += 256) dis[i] = 1e10f;
  if (t == 0) { idx[(long)blockIdx.x * m] = 0; s_last = 0; }
  __syncthreads();
  for (int it = 1; it < m; ++it) {
    int last = s_last;
    float px = base[last * 3], py = base[last * 3 + 1], pz = base[last * 3 + 2];
    float best = -1.f; int barg = n;
    for (int i = t; i < n; i += 256) {
      float dx = base[i * 3] - px, dy = base[i * 3 + 1] - py, dz = base[i * 3 + 2] - pz;
      float nd = fminf(dis[i], dx * dx + dy * dy + dz * dz);
      dis[i] = nd;
      if (nd > best) { best = nd; barg = i; }
    }
    rv[t] = best; ri[t] = barg;
    __syncthreads();
    for (int s = 128; s > 0; s >>= 1) {
      if (t < s) {
        float v2 = rv[t + s]; int i2 = ri[t + s];
        if (v2 > rv[t] || (v2 == rv[t] && i2 < ri[t])) { rv[t] = v2; ri[t] = i2; }
      }
      __syncthreads();
    }
    if (t == 0) { s_last = ri[0]; idx[(long)blockIdx.x * m + it] = ri[0]; }
    __syncthreads();
  }
}

__global__ void k_gather3(const float* __restrict__ pts, const int* __restrict__ idx,
                          float* __restrict__ cent, int n, int m, int total) {
  int i = blockIdx.x * blockDim.x + threadIdx.x;
  if (i >= total) return;                    // total = B*m
  int b_i = i / m, mi = i % m;
  int s = idx[(long)b_i * m + mi];
#pragma unroll
  for (int c = 0; c < 3; ++c)
    cent[((long)b_i * m + mi) * 3 + c] = pts[((long)b_i * n + s) * 3 + c];
}

// ball query: 32 smallest in-radius, out-of-radius slots padded with slot 0.
__global__ void k_ballquery(const float* __restrict__ cent, const float* __restrict__ pts,
                            int n, int m, float r2, int* __restrict__ out) {
  extern __shared__ float dist[];
  __shared__ float rv[256];
  __shared__ int ri[256];
  __shared__ int firstIdx;
  int blk = blockIdx.x;
  int b_i = blk / m;
  const float* base = pts + (long)b_i * n * 3;
  float cx = cent[(long)blk * 3], cy = cent[(long)blk * 3 + 1], cz = cent[(long)blk * 3 + 2];
  int t = threadIdx.x;
  for (int i = t; i < n; i += 256) {
    float dx = base[i * 3] - cx, dy = base[i * 3 + 1] - cy, dz = base[i * 3 + 2] - cz;
    float d = dx * dx + dy * dy + dz * dz;
    dist[i] = (d <= r2) ? d : FINF;
  }
  __syncthreads();
  for (int kk = 0; kk < KNB; ++kk) {
    float best = FINF; int barg = n;
    for (int i = t; i < n; i += 256) {
      float d = dist[i];
      if (d < best || (d == best && i < barg)) { best = d; barg = i; }
    }
    rv[t] = best; ri[t] = barg;
    __syncthreads();
    for (int s = 128; s > 0; s >>= 1) {
      if (t < s) {
        if (rv[t + s] < rv[t] || (rv[t + s] == rv[t] && ri[t + s] < ri[t])) {
          rv[t] = rv[t + s]; ri[t] = ri[t + s];
        }
      }
      __syncthreads();
    }
    if (t == 0) {
      int sel = (ri[0] < n) ? ri[0] : 0;
      if (kk == 0) firstIdx = sel;
      if (rv[0] >= FINF) sel = firstIdx;
      out[(long)blk * KNB + kk] = sel;
      if (ri[0] < n) dist[ri[0]] = FINF;
    }
    __syncthreads();
  }
}

// H[(b,m,kk), :] = [ (p[idx]-cent)/r , feats[idx] , 0-pad ]  (bf16)
__global__ void k_build_h(const float* __restrict__ pts, const float* __restrict__ cent,
                          const float* __restrict__ feats, const int* __restrict__ bq,
                          unsigned short* __restrict__ H, int n, int m, int C, int Kpad,
                          float inv_r, long total) {
  long i = (long)blockIdx.x * blockDim.x + threadIdx.x;
  if (i >= total) return;
  int col = (int)(i % Kpad);
  long row = i / Kpad;
  int b_i = (int)(row / ((long)m * KNB));
  int rem = (int)(row % ((long)m * KNB));
  int mi = rem >> 5;
  int src = bq[row];
  float val;
  if (col < 3)
    val = (pts[((long)b_i * n + src) * 3 + col] - cent[((long)b_i * m + mi) * 3 + col]) * inv_r;
  else if (col < 3 + C)
    val = feats[((long)b_i * n + src) * C + (col - 3)];
  else
    val = 0.f;
  H[i] = f2bf(val);
}

// fp32 (M,N) -> bf16 (M,N), plain cast (activations, row-major A operand).
__global__ void k_cast_bf(const float* __restrict__ src, unsigned short* __restrict__ dst,
                          long total) {
  long i = (long)blockIdx.x * blockDim.x + threadIdx.x;
  if (i >= total) return;
  dst[i] = f2bf(src[i]);
}

// weight fp32 (K,N) -> TRANSPOSED bf16 (N, Kpad) with zero K padding.
__global__ void k_pad_wT(const float* __restrict__ src, unsigned short* __restrict__ dst,
                         int K, int Kpad, int Ncols, long total) {
  long i = (long)blockIdx.x * blockDim.x + threadIdx.x;
  if (i >= total) return;                    // total = Ncols*Kpad
  int kp = (int)(i % Kpad);
  long ncol = i / Kpad;
  dst[i] = (kp < K) ? f2bf(src[(long)kp * Ncols + ncol]) : (unsigned short)0;
}

// 3-NN inverse-distance weights (n2 small)
__global__ void k_three_nn(const float* __restrict__ p1, const float* __restrict__ p2,
                           int n1, int n2, int* __restrict__ idx3, float* __restrict__ w3, int total) {
  int i = blockIdx.x * blockDim.x + threadIdx.x;
  if (i >= total) return;                     // total = B*n1
  int b_i = i / n1;
  const float* a = p1 + (long)i * 3;
  const float* base = p2 + (long)b_i * n2 * 3;
  float d0 = FINF, d1 = FINF, d2 = FINF;
  int i0 = 0, i1 = 0, i2 = 0;
  for (int j = 0; j < n2; ++j) {
    float dx = a[0] - base[j * 3], dy = a[1] - base[j * 3 + 1], dz = a[2] - base[j * 3 + 2];
    float d = dx * dx + dy * dy + dz * dz;
    if (d < d0)      { d2 = d1; i2 = i1; d1 = d0; i1 = i0; d0 = d; i0 = j; }
    else if (d < d1) { d2 = d1; i2 = i1; d1 = d;  i1 = j; }
    else if (d < d2) { d2 = d;  i2 = j; }
  }
  float r0 = 1.f / (d0 + 1e-8f), r1 = 1.f / (d1 + 1e-8f), r2 = 1.f / (d2 + 1e-8f);
  float s = r0 + r1 + r2;
  w3[(long)i * 3 + 0] = r0 / s; w3[(long)i * 3 + 1] = r1 / s; w3[(long)i * 3 + 2] = r2 / s;
  idx3[(long)i * 3 + 0] = i0;   idx3[(long)i * 3 + 1] = i1;   idx3[(long)i * 3 + 2] = i2;
}

// H_fp[(b,i), :] = [ sum_j w*f2[idx_j] (1024) , f1 (512) ]  bf16
__global__ void k_build_fp(const float* __restrict__ f2, const float* __restrict__ f1,
                           const int* __restrict__ idx3, const float* __restrict__ w3,
                           unsigned short* __restrict__ H, long total) {
  long i = (long)blockIdx.x * blockDim.x + threadIdx.x;
  if (i >= total) return;
  int col = (int)(i % 1536);
  long row = i / 1536;
  int b_i = (int)(row / 128);
  float val;
  if (col < 1024) {
    const int* id = idx3 + row * 3;
    const float* ww = w3 + row * 3;
    val = ww[0] * f2[((long)b_i * 32 + id[0]) * 1024 + col]
        + ww[1] * f2[((long)b_i * 32 + id[1]) * 1024 + col]
        + ww[2] * f2[((long)b_i * 32 + id[2]) * 1024 + col];
  } else {
    val = f1[row * 512 + (col - 1024)];
  }
  H[i] = f2bf(val);
}

// out(b, 13, m) = swapaxes(h @ w2 + b2)
__global__ void k_head(const float* __restrict__ h, const float* __restrict__ w2,
                       const float* __restrict__ b2, float* __restrict__ out,
                       int m, int Kc, int total) {
  int i = blockIdx.x * blockDim.x + threadIdx.x;
  if (i >= total) return;                     // total = B*13*m
  int mi = i % m;
  int rem = i / m;
  int cls = rem % 13;
  int b_i = rem / 13;
  const float* hr = h + ((long)b_i * m + mi) * Kc;
  float acc = b2[cls];
  for (int k = 0; k < Kc; ++k) acc += hr[k] * w2[(long)k * 13 + cls];
  out[i] = acc;
}

__global__ void k_copy(const float* __restrict__ src, float* __restrict__ dst, long total) {
  long i = (long)blockIdx.x * blockDim.x + threadIdx.x;
  if (i < total) dst[i] = src[i];
}

// ---------------------------------------------------------------------------
// Host orchestration
// ---------------------------------------------------------------------------
static inline long cdivl(long a, long b) { return (a + b - 1) / b; }

extern "C" void kernel_launch(void* const* d_in, const int* in_sizes, int n_in,
                              void* d_out, int out_size, void* d_ws, size_t ws_size,
                              hipStream_t stream) {
  (void)in_sizes; (void)n_in; (void)out_size; (void)ws_size;

  // Input layout (setup_inputs dict order, nested dicts/lists flattened in
  // insertion order):
  //   0 points (4,8192,3)   1 feat (4,8192,4)   2 in_w (4,64)   3 in_b (64)
  //   stage i: base 4+12i -> sa{w,g,b}, inv{w1,g1,b1,w2a,g2a,b2a,w2b,g2b,b2b}
  //   52..56 cm1{w1,g1,b1,w2,b2}  57..62 fp1{w1,g1,b1,w2,g2,b2}
  //   63..67 cm2{w1,g1,b1,w2,b2}
  const float* points = (const float*)d_in[0];
  const float* feat   = (const float*)d_in[1];
  const float* in_w   = (const float*)d_in[2];
  const float* in_b   = (const float*)d_in[3];

  const int ns[5]    = {8192, 2048, 512, 128, 32};
  const int cins[4]  = {64, 128, 256, 512};
  const int couts[4] = {128, 256, 512, 1024};
  const float radii[4] = {0.1f, 0.2f, 0.4f, 0.8f};

  char* ws = (char*)d_ws;
  size_t off = 0;
  auto alloc = [&](size_t bytes) -> void* {
    off = (off + 255) & ~(size_t)255;
    void* p = ws + off;
    off += bytes;
    return p;
  };

  // persistent buffers -----------------------------------------------------
  float* x_s[5];
  const float* pos_s[5];
  pos_s[0] = points;
  x_s[0] = (float*)alloc((size_t)BB * N0 * 64 * 4);
  for (int i = 0; i < 4; ++i) {
    pos_s[i + 1] = (float*)alloc((size_t)BB * ns[i + 1] * 3 * 4);
    x_s[i + 1]   = (float*)alloc((size_t)BB * ns[i + 1] * couts[i] * 4);
  }
  int* fpsidx = (int*)alloc((size_t)BB * 2048 * 4);
  int* bq     = (int*)alloc((size_t)BB * 2048 * KNB * 4);
  float* x3f  = (float*)alloc((size_t)BB * 128 * 512 * 4);   // fp output
  size_t scratch_mark = off;

  // in_linear --------------------------------------------------------------
  {
    long tot = (long)BB * N0 * 64;
    k_in_linear<<<cdivl(tot, 256), 256, 0, stream>>>(feat, in_w, in_b, x_s[0], tot);
  }

  // stages -----------------------------------------------------------------
  for (int i = 0; i < 4; ++i) {
    int n = ns[i], m = ns[i + 1], cin = cins[i], cout = couts[i];
    float r = radii[i];
    int pb = 4 + 12 * i;
    const float* sa_w  = (const float*)d_in[pb + 0];
    const float* sa_g  = (const float*)d_in[pb + 1];
    const float* sa_b  = (const float*)d_in[pb + 2];
    const float* i_w1  = (const float*)d_in[pb + 3];
    const float* i_g1  = (const float*)d_in[pb + 4];
    const float* i_b1  = (const float*)d_in[pb + 5];
    const float* i_w2a = (const float*)d_in[pb + 6];
    const float* i_g2a = (const float*)d_in[pb + 7];
    const float* i_b2a = (const float*)d_in[pb + 8];
    const float* i_w2b = (const float*)d_in[pb + 9];
    const float* i_g2b = (const float*)d_in[pb + 10];
    const float* i_b2b = (const float*)d_in[pb + 11];

    // ---- SA layer: FPS -> ball query -> gather -> WMMA GEMM + maxpool ----
    k_fps<<<BB, 256, n * 4, stream>>>(pos_s[i], n, m, fpsidx);
    k_gather3<<<cdivl((long)BB * m, 256), 256, 0, stream>>>(pos_s[i], fpsidx,
                                                            (float*)pos_s[i + 1], n, m, BB * m);
    k_ballquery<<<BB * m, 256, n * 4, stream>>>(pos_s[i + 1], pos_s[i], n, m, r * r, bq);

    off = scratch_mark;
    int d = cin + 3, Kpad = (d + 31) & ~31;
    long rows = (long)BB * m * KNB;
    unsigned short* H = (unsigned short*)alloc((size_t)rows * Kpad * 2);
    k_build_h<<<cdivl(rows * Kpad, 256), 256, 0, stream>>>(
        pos_s[i], pos_s[i + 1], x_s[i], bq, H, n, m, cin, Kpad, 1.f / r, rows * Kpad);
    unsigned short* Wb = (unsigned short*)alloc((size_t)cout * Kpad * 2);
    k_pad_wT<<<cdivl((long)cout * Kpad, 256), 256, 0, stream>>>(sa_w, Wb, d, Kpad, cout,
                                                                (long)cout * Kpad);
    k_gemm_pool<<<dim3(BB * m, cout / 128), 256, 0, stream>>>(H, Wb, sa_g, sa_b,
                                                              x_s[i + 1], Kpad, cout);

    // ---- inverted residual block (radius*2) ----
    float r2x = 2.f * r;
    k_ballquery<<<BB * m, 256, m * 4, stream>>>(pos_s[i + 1], pos_s[i + 1], m, m,
                                                r2x * r2x, bq);
    off = scratch_mark;
    int d2 = cout + 3, Kpad2 = (d2 + 31) & ~31;
    long rows2 = (long)BB * m * KNB;
    unsigned short* H2 = (unsigned short*)alloc((size_t)rows2 * Kpad2 * 2);
    k_build_h<<<cdivl(rows2 * Kpad2, 256), 256, 0, stream>>>(
        pos_s[i + 1], pos_s[i + 1], x_s[i + 1], bq, H2, m, m, cout, Kpad2, 1.f / r2x,
        rows2 * Kpad2);
    unsigned short* Wb1 = (unsigned short*)alloc((size_t)cout * Kpad2 * 2);
    k_pad_wT<<<cdivl((long)cout * Kpad2, 256), 256, 0, stream>>>(i_w1, Wb1, d2, Kpad2, cout,
                                                                 (long)cout * Kpad2);
    long Mrows = (long)BB * m;                       // 8192 / 2048 / 512 / 128
    float* hpool = (float*)alloc((size_t)Mrows * cout * 4);
    k_gemm_pool<<<dim3(BB * m, cout / 128), 256, 0, stream>>>(H2, Wb1, i_g1, i_b1,
                                                              hpool, Kpad2, cout);
    // h = relu(bn(h @ w2a))      (cout -> 4*cout)
    int c4 = 4 * cout;
    unsigned short* hbf = (unsigned short*)alloc((size_t)Mrows * cout * 2);
    k_cast_bf<<<cdivl(Mrows * cout, 256), 256, 0, stream>>>(hpool, hbf, Mrows * cout);
    unsigned short* W2ab = (unsigned short*)alloc((size_t)c4 * cout * 2);
    k_pad_wT<<<cdivl((long)c4 * cout, 256), 256, 0, stream>>>(i_w2a, W2ab, cout, cout, c4,
                                                              (long)c4 * cout);
    float* h4 = (float*)alloc((size_t)Mrows * c4 * 4);
    k_gemm<<<dim3(Mrows / 128, c4 / 64), 256, 0, stream>>>(hbf, W2ab, i_g2a, i_b2a,
                                                           nullptr, h4, cout, c4, 1);
    // x = relu(x + bn(h @ w2b))  (4*cout -> cout), in-place residual
    unsigned short* h4bf = (unsigned short*)alloc((size_t)Mrows * c4 * 2);
    k_cast_bf<<<cdivl(Mrows * c4, 256), 256, 0, stream>>>(h4, h4bf, Mrows * c4);
    unsigned short* W2bb = (unsigned short*)alloc((size_t)cout * c4 * 2);
    k_pad_wT<<<cdivl((long)cout * c4, 256), 256, 0, stream>>>(i_w2b, W2bb, c4, c4, cout,
                                                              (long)cout * c4);
    k_gemm<<<dim3(Mrows / 128, cout / 64), 256, 0, stream>>>(h4bf, W2bb, i_g2b, i_b2b,
                                                             x_s[i + 1], x_s[i + 1], c4, cout, 1);
    off = scratch_mark;
  }

  // heads + FP layer -------------------------------------------------------
  const float* pos3 = pos_s[3];  float* x3 = x_s[3];   // (4,128,512)
  const float* pos4 = pos_s[4];  float* x4 = x_s[4];   // (4,32,1024)
  float* out = (float*)d_out;

  const float* cm1_w1 = (const float*)d_in[52];
  const float* cm1_g1 = (const float*)d_in[53];
  const float* cm1_b1 = (const float*)d_in[54];
  const float* cm1_w2 = (const float*)d_in[55];
  const float* cm1_b2 = (const float*)d_in[56];
  const float* fp_w1  = (const float*)d_in[57];
  const float* fp_g1  = (const float*)d_in[58];
  const float* fp_b1  = (const float*)d_in[59];
  const float* fp_w2  = (const float*)d_in[60];
  const float* fp_g2  = (const float*)d_in[61];
  const float* fp_b2  = (const float*)d_in[62];
  const float* cm2_w1 = (const float*)d_in[63];
  const float* cm2_g1 = (const float*)d_in[64];
  const float* cm2_b1 = (const float*)d_in[65];
  const float* cm2_w2 = (const float*)d_in[66];
  const float* cm2_b2 = (const float*)d_in[67];

  off = scratch_mark;

  // coarse1 = head(x4): 128x1024 @ 1024x1024 -> relu(bn) -> @1024x13 + b
  {
    long M = (long)BB * 32;
    unsigned short* xbf = (unsigned short*)alloc((size_t)M * 1024 * 2);
    k_cast_bf<<<cdivl(M * 1024, 256), 256, 0, stream>>>(x4, xbf, M * 1024);
    unsigned short* Wc = (unsigned short*)alloc((size_t)1024 * 1024 * 2);
    k_pad_wT<<<cdivl(1024L * 1024, 256), 256, 0, stream>>>(cm1_w1, Wc, 1024, 1024, 1024,
                                                           1024L * 1024);
    float* h1 = (float*)alloc((size_t)M * 1024 * 4);
    k_gemm<<<dim3(M / 128, 1024 / 64), 256, 0, stream>>>(xbf, Wc, cm1_g1, cm1_b1,
                                                         nullptr, h1, 1024, 1024, 1);
    k_head<<<cdivl((long)BB * 13 * 32, 256), 256, 0, stream>>>(h1, cm1_w2, cm1_b2,
                                                               out + 0, 32, 1024, BB * 13 * 32);
  }

  // fp_layer: 3-NN interp (pos3 <- pos4), concat, two WMMA GEMMs
  {
    long M = (long)BB * 128;
    int* idx3 = (int*)alloc((size_t)M * 3 * 4);
    float* w3 = (float*)alloc((size_t)M * 3 * 4);
    k_three_nn<<<cdivl(M, 256), 256, 0, stream>>>(pos3, pos4, 128, 32, idx3, w3, (int)M);
    unsigned short* Hfp = (unsigned short*)alloc((size_t)M * 1536 * 2);
    k_build_fp<<<cdivl(M * 1536, 256), 256, 0, stream>>>(x4, x3, idx3, w3, Hfp, M * 1536);
    unsigned short* W1 = (unsigned short*)alloc((size_t)512 * 1536 * 2);
    k_pad_wT<<<cdivl(512L * 1536, 256), 256, 0, stream>>>(fp_w1, W1, 1536, 1536, 512,
                                                          512L * 1536);
    float* t1 = (float*)alloc((size_t)M * 512 * 4);
    k_gemm<<<dim3(M / 128, 512 / 64), 256, 0, stream>>>(Hfp, W1, fp_g1, fp_b1,
                                                        nullptr, t1, 1536, 512, 1);
    unsigned short* t1bf = (unsigned short*)alloc((size_t)M * 512 * 2);
    k_cast_bf<<<cdivl(M * 512, 256), 256, 0, stream>>>(t1, t1bf, M * 512);
    unsigned short* W2 = (unsigned short*)alloc((size_t)512 * 512 * 2);
    k_pad_wT<<<cdivl(512L * 512, 256), 256, 0, stream>>>(fp_w2, W2, 512, 512, 512,
                                                         512L * 512);
    k_gemm<<<dim3(M / 128, 512 / 64), 256, 0, stream>>>(t1bf, W2, fp_g2, fp_b2,
                                                        nullptr, x3f, 512, 512, 1);
  }

  // coarse2 = head(x3f): 512x512 @ 512x512 -> relu(bn) -> @512x13 + b
  {
    long M = (long)BB * 128;
    unsigned short* xbf = (unsigned short*)alloc((size_t)M * 512 * 2);
    k_cast_bf<<<cdivl(M * 512, 256), 256, 0, stream>>>(x3f, xbf, M * 512);
    unsigned short* Wc = (unsigned short*)alloc((size_t)512 * 512 * 2);
    k_pad_wT<<<cdivl(512L * 512, 256), 256, 0, stream>>>(cm2_w1, Wc, 512, 512, 512,
                                                         512L * 512);
    float* h2 = (float*)alloc((size_t)M * 512 * 4);
    k_gemm<<<dim3(M / 128, 512 / 64), 256, 0, stream>>>(xbf, Wc, cm2_g1, cm2_b1,
                                                        nullptr, h2, 512, 512, 1);
    k_head<<<cdivl((long)BB * 13 * 128, 256), 256, 0, stream>>>(h2, cm2_w2, cm2_b2,
                                                                out + 1664, 128, 512,
                                                                BB * 13 * 128);
  }

  // x3f -> d_out tail
  k_copy<<<cdivl(262144L, 256), 256, 0, stream>>>(x3f, out + 1664 + 6656, 262144L);
}
